// EvolveGCNH_78065325572380
// MI455X (gfx1250) — matrix operand order repaired and checked by reading, not compiled, not run on hardware
//
#include <hip/hip_runtime.h>
#include <hip/hip_bf16.h>
#include <math.h>

// ---------------- problem constants ----------------
#define Bq   64
#define Nq   512
#define Eq   16384
#define INq  64
#define HIDq 64
#define OUTq 2
#define H0q  4160   // HID*IN + HID
#define H1q  130    // OUT*HID + OUT
#define GINq 4096   // TOPK*IN

// ---------------- workspace layout (float-sized slots) ----------------
constexpr size_t O_HINIT0 = 0;
constexpr size_t O_HINIT1 = O_HINIT0 + H0q;                    // 4160
constexpr size_t O_GH0    = O_HINIT1 + H1q;                    // 4290
constexpr size_t O_GH1    = O_GH0 + 3*H0q;                     // + 12480
constexpr size_t O_ZBF    = O_GH1 + 3*H1q;                     // bf16: Bq*GINq elems = 131072 floats
constexpr size_t O_GI0    = O_ZBF + (size_t)Bq*GINq/2;
constexpr size_t O_HN0    = O_GI0 + (size_t)3*H0q*Bq;          // + 798720
constexpr size_t O_WBF    = O_HN0 + (size_t)Bq*H0q;            // bf16: Bq*4096 elems = 131072 floats
constexpr size_t O_DINV   = O_WBF + (size_t)Bq*GINq/2;
constexpr size_t O_XW1    = O_DINV + (size_t)Bq*Nq;
constexpr size_t O_AGG1   = O_XW1 + (size_t)Bq*Nq*HIDq;
constexpr size_t O_GI1    = O_AGG1 + (size_t)Bq*Nq*HIDq;
constexpr size_t O_HN1    = O_GI1 + (size_t)3*H1q*Bq;
constexpr size_t O_XW2    = O_HN1 + (size_t)Bq*H1q;
// total ~ 5.6M floats ~ 22.5 MB

// ---------------- WMMA types ----------------
typedef __attribute__((ext_vector_type(16))) __bf16 v16bf;
typedef __attribute__((ext_vector_type(8)))  __bf16 v8bf;
typedef __attribute__((ext_vector_type(8)))  float  v8f;

__device__ __forceinline__ __bf16 f2bf(float f) { return (__bf16)f; }

// A fragment (f32 source, converted in-register): rows of row-major matrix, K-slice [k0,k0+32)
// e0..7 <- K = k0+half*8 .. +7 ; e8..15 <- K = k0+16+half*8 .. +7
__device__ __forceinline__ v16bf load_afrag(const float* __restrict__ arow, int k0, int half) {
  v16bf a;
  const float4* p0 = reinterpret_cast<const float4*>(arow + k0 + half*8);
  const float4* p1 = reinterpret_cast<const float4*>(arow + k0 + 16 + half*8);
  float4 x0 = p0[0], x1 = p0[1], y0 = p1[0], y1 = p1[1];
  a[0]=f2bf(x0.x); a[1]=f2bf(x0.y); a[2]=f2bf(x0.z); a[3]=f2bf(x0.w);
  a[4]=f2bf(x1.x); a[5]=f2bf(x1.y); a[6]=f2bf(x1.z); a[7]=f2bf(x1.w);
  a[8]=f2bf(y0.x); a[9]=f2bf(y0.y); a[10]=f2bf(y0.z); a[11]=f2bf(y0.w);
  a[12]=f2bf(y1.x); a[13]=f2bf(y1.y); a[14]=f2bf(y1.z); a[15]=f2bf(y1.w);
  return a;
}

// B fragment (pre-converted bf16 source): B(k,n)=Brow_n[k], K-slice [k0,k0+32)
// e0..7 <- K = k0+half*8 .. +7 ; e8..15 <- K = k0+16+half*8 .. +7 => 2x16B loads, no cvt
__device__ __forceinline__ v16bf load_bfrag_bf(const __bf16* __restrict__ brow, int k0, int half) {
  v8bf x0 = *reinterpret_cast<const v8bf*>(brow + k0 + half*8);
  v8bf x1 = *reinterpret_cast<const v8bf*>(brow + k0 + 16 + half*8);
  return __builtin_shufflevector(x0, x1, 0,1,2,3,4,5,6,7,8,9,10,11,12,13,14,15);
}

// ==========================================================
// K1: h_init0 = [W0.ravel(), b0], h_init1 = [W1.ravel(), b1]
// ==========================================================
__global__ void k_build_hinit(const float* __restrict__ W0, const float* __restrict__ b0,
                              const float* __restrict__ W1, const float* __restrict__ b1,
                              float* __restrict__ h0, float* __restrict__ h1) {
  int t = blockIdx.x * blockDim.x + threadIdx.x;
  if (t < H0q) h0[t] = (t < HIDq*INq) ? W0[t] : b0[t - HIDq*INq];
  if (t < H1q) h1[t] = (t < OUTq*HIDq) ? W1[t] : b1[t - OUTq*HIDq];
}

// ==========================================================
// K2: out[m] = dot(W[m,:], v) + bias[m]   (one wave32 per row)
// ==========================================================
__global__ void k_matvec_rows(const float* __restrict__ W, const float* __restrict__ v,
                              const float* __restrict__ bias, float* __restrict__ out,
                              int M, int K) {
  int wave = (int)((blockIdx.x * blockDim.x + threadIdx.x) >> 5);
  int lane = threadIdx.x & 31;
  if (wave >= M) return;
  const float* row = W + (size_t)wave * K;
  float acc = 0.f;
  for (int k = lane; k < K; k += 32) acc += row[k] * v[k];
  for (int off = 16; off > 0; off >>= 1) acc += __shfl_down(acc, off, 32);
  if (lane == 0) out[wave] = acc + bias[wave];
}

// ==========================================================
// K3: summarize: y = h@(p/|p|); top-64 (desc, stable);
//     Z[b, k] = bf16( h[sel[k/64], k%64] * tanh(y[sel[k/64]]) )
// ==========================================================
__global__ void k_summarize(const float* __restrict__ h, const float* __restrict__ p,
                            __bf16* __restrict__ Z) {
  int b = blockIdx.x;
  int t = threadIdx.x;                 // 0..511
  __shared__ float sy[Nq];
  __shared__ float sty[Nq];
  __shared__ float rmax[Nq];
  __shared__ int   ridx[Nq];
  __shared__ float spn[INq];
  __shared__ int   ssel[64];
  __shared__ float snorm;

  if (t == 0) {
    float s = 0.f;
    for (int j = 0; j < INq; ++j) s += p[j] * p[j];
    snorm = sqrtf(s) + 1e-8f;
  }
  __syncthreads();
  if (t < INq) spn[t] = p[t] / snorm;
  __syncthreads();

  const float* hb = h + (size_t)b * Nq * INq;
  const float* hr = hb + (size_t)t * INq;
  float y = 0.f;
  for (int j = 0; j < INq; ++j) y += hr[j] * spn[j];
  sy[t]  = y;
  sty[t] = tanhf(y);
  __syncthreads();

  for (int r = 0; r < 64; ++r) {
    rmax[t] = sy[t];
    ridx[t] = t;
    __syncthreads();
    for (int s = 256; s > 0; s >>= 1) {
      if (t < s) {
        float vo = rmax[t + s]; int io = ridx[t + s];
        if (vo > rmax[t] || (vo == rmax[t] && io < ridx[t])) { rmax[t] = vo; ridx[t] = io; }
      }
      __syncthreads();
    }
    if (t == 0) { ssel[r] = ridx[0]; sy[ridx[0]] = -3.4e38f; }
    __syncthreads();
  }

  __bf16* Zb = Z + (size_t)b * GINq;
  for (int k = t; k < GINq; k += Nq) {
    int r = k >> 6, j = k & 63;
    int idx = ssel[r];
    Zb[k] = f2bf(hb[(size_t)idx * INq + j] * sty[idx]);
  }
}

// ==========================================================
// K4: WMMA GEMM: C[m, b] = sum_k A[m,k] * Z[b,k] + bias[m]
//     A: M x 4096 f32 (read ONCE, cvt in-register), Z: 64 x 4096 bf16
//     block = 256 (8 waves); each wave: one 16-row tile x full N=64
//     K loop unrolled x2 so the scheduler can overlap the next
//     step's B loads with this step's WMMAs (distinct registers)
// ==========================================================
__global__ void k_gemm_wmma(const float* __restrict__ A, const __bf16* __restrict__ Z,
                            const float* __restrict__ bias, float* __restrict__ C, int M) {
  const int K = GINq;
  int wave = threadIdx.x >> 5;
  int lane = threadIdx.x & 31;
  int mTiles = (M + 15) >> 4;
  int mTile = blockIdx.x * 8 + wave;
  if (mTile >= mTiles) return;       // wave-uniform: EXEC stays all-ones
  int half = lane >> 4;
  int l16  = lane & 15;
  int rowA = mTile * 16 + l16; if (rowA >= M) rowA = M - 1;
  const float* arow = A + (size_t)rowA * K;
  const __bf16* brow0 = Z + (size_t)(0  + l16) * K;
  const __bf16* brow1 = Z + (size_t)(16 + l16) * K;
  const __bf16* brow2 = Z + (size_t)(32 + l16) * K;
  const __bf16* brow3 = Z + (size_t)(48 + l16) * K;

  v8f c0, c1, c2, c3;
#pragma unroll
  for (int i = 0; i < 8; ++i) { c0[i]=0.f; c1[i]=0.f; c2[i]=0.f; c3[i]=0.f; }

#pragma unroll 2
  for (int k0 = 0; k0 < K; k0 += 32) {
    __builtin_prefetch(arow + k0 + 512, 0, 0);   // stream A ahead
    // load everything first (distinct values), then 4 back-to-back WMMAs
    v16bf a  = load_afrag(arow, k0, half);
    v16bf b0 = load_bfrag_bf(brow0, k0, half);
    v16bf b1 = load_bfrag_bf(brow1, k0, half);
    v16bf b2 = load_bfrag_bf(brow2, k0, half);
    v16bf b3 = load_bfrag_bf(brow3, k0, half);
    c0 = __builtin_amdgcn_wmma_f32_16x16x32_bf16(false, a, false, b0, (short)0, c0, false, false);
    c1 = __builtin_amdgcn_wmma_f32_16x16x32_bf16(false, a, false, b1, (short)0, c1, false, false);
    c2 = __builtin_amdgcn_wmma_f32_16x16x32_bf16(false, a, false, b2, (short)0, c2, false, false);
    c3 = __builtin_amdgcn_wmma_f32_16x16x32_bf16(false, a, false, b3, (short)0, c3, false, false);
  }
#pragma unroll
  for (int r = 0; r < 8; ++r) {
    int m = mTile * 16 + r + half * 8;
    if (m < M) {
      float bs = bias[m];
      float* crow = C + (size_t)m * Bq;
      crow[0  + l16] = c0[r] + bs;
      crow[16 + l16] = c1[r] + bs;
      crow[32 + l16] = c2[r] + bs;
      crow[48 + l16] = c3[r] + bs;
    }
  }
}

// ==========================================================
// K5: GRU combine; optionally emits bf16 copy of the evolved
//     weight block (first Wcnt entries) for the WMMA consumer
// ==========================================================
__global__ void k_gru_combine(const float* __restrict__ GI, const float* __restrict__ gh,
                              const float* __restrict__ hinit, float* __restrict__ hn,
                              __bf16* __restrict__ wbf, int H, int Wcnt) {
  int idx = blockIdx.x * blockDim.x + threadIdx.x;
  if (idx >= Bq * H) return;
  int b = idx & (Bq - 1);
  int m = idx >> 6;
  float gir = GI[(size_t)m * Bq + b];
  float giz = GI[(size_t)(H + m) * Bq + b];
  float gin = GI[(size_t)(2*H + m) * Bq + b];
  float r = 1.f / (1.f + expf(-(gir + gh[m])));
  float z = 1.f / (1.f + expf(-(giz + gh[H + m])));
  float n = tanhf(gin + r * gh[2*H + m]);
  float val = (1.f - z) * n + z * hinit[m];
  hn[(size_t)b * H + m] = val;
  if (wbf != nullptr && m < Wcnt) wbf[(size_t)b * Wcnt + m] = f2bf(val);
}

// ==========================================================
// K6: degree / dinv
// ==========================================================
__global__ void k_deg_init(float* __restrict__ deg) {
  int i = blockIdx.x * blockDim.x + threadIdx.x;
  if (i < Bq * Nq) deg[i] = 1.0f;   // self-loop weight
}
__global__ void k_deg_edges(const int* __restrict__ ei, const float* __restrict__ ew,
                            float* __restrict__ deg) {
  int idx = blockIdx.x * blockDim.x + threadIdx.x;
  if (idx >= Bq * Eq) return;
  int b = idx / Eq, e = idx % Eq;
  int col = ei[(size_t)b * 2 * Eq + Eq + e];
  atomicAdd(&deg[(size_t)b * Nq + col], ew[(size_t)b * Eq + e]);
}
__global__ void k_deg_finalize(float* __restrict__ deg) {
  int i = blockIdx.x * blockDim.x + threadIdx.x;
  if (i < Bq * Nq) deg[i] = rsqrtf(deg[i]);   // deg >= 1 always
}

// ==========================================================
// K7: layer-1 xw = x[b] @ Wn[b]^T via WMMA; weights pre-cvt bf16
// ==========================================================
__global__ void k_xw1_wmma(const float* __restrict__ X, const __bf16* __restrict__ wbf,
                           float* __restrict__ XW) {
  int b = blockIdx.x;
  int wave = threadIdx.x >> 5;
  int lane = threadIdx.x & 31;
  int mTile = blockIdx.y * 8 + wave;   // 0..31
  int half = lane >> 4;
  int l16  = lane & 15;
  const float* arow = X + ((size_t)b * Nq + mTile * 16 + l16) * INq;
  const __bf16* wbase = wbf + (size_t)b * GINq;
  const __bf16* brow0 = wbase + (size_t)(0  + l16) * HIDq;
  const __bf16* brow1 = wbase + (size_t)(16 + l16) * HIDq;
  const __bf16* brow2 = wbase + (size_t)(32 + l16) * HIDq;
  const __bf16* brow3 = wbase + (size_t)(48 + l16) * HIDq;

  v8f c0, c1, c2, c3;
#pragma unroll
  for (int i = 0; i < 8; ++i) { c0[i]=0.f; c1[i]=0.f; c2[i]=0.f; c3[i]=0.f; }

#pragma unroll
  for (int k0 = 0; k0 < INq; k0 += 32) {
    v16bf a  = load_afrag(arow, k0, half);
    v16bf b0 = load_bfrag_bf(brow0, k0, half);
    v16bf b1 = load_bfrag_bf(brow1, k0, half);
    v16bf b2 = load_bfrag_bf(brow2, k0, half);
    v16bf b3 = load_bfrag_bf(brow3, k0, half);
    c0 = __builtin_amdgcn_wmma_f32_16x16x32_bf16(false, a, false, b0, (short)0, c0, false, false);
    c1 = __builtin_amdgcn_wmma_f32_16x16x32_bf16(false, a, false, b1, (short)0, c1, false, false);
    c2 = __builtin_amdgcn_wmma_f32_16x16x32_bf16(false, a, false, b2, (short)0, c2, false, false);
    c3 = __builtin_amdgcn_wmma_f32_16x16x32_bf16(false, a, false, b3, (short)0, c3, false, false);
  }
#pragma unroll
  for (int r = 0; r < 8; ++r) {
    int m = mTile * 16 + r + half * 8;
    float* crow = XW + ((size_t)b * Nq + m) * HIDq;
    crow[0  + l16] = c0[r];
    crow[16 + l16] = c1[r];
    crow[32 + l16] = c2[r];
    crow[48 + l16] = c3[r];
  }
}

// ==========================================================
// K8: agg init with self loop: agg[b,i,c] = xw[b,i,c]*dinv[b,i]^2
// ==========================================================
__global__ void k_agg_init64(const float* __restrict__ xw, const float* __restrict__ dinv,
                             float* __restrict__ agg) {
  int idx = blockIdx.x * blockDim.x + threadIdx.x;
  if (idx >= Bq * Nq * HIDq) return;
  int node = idx >> 6;     // b*Nq + i
  float d = dinv[node];
  agg[idx] = xw[idx] * d * d;
}

// ==========================================================
// K9: edge scatter (64 channels): 4 threads/edge, 16 ch each
// ==========================================================
__global__ void k_edge_scatter64(const int* __restrict__ ei, const float* __restrict__ ew,
                                 const float* __restrict__ dinv, const float* __restrict__ xw,
                                 float* __restrict__ agg) {
  long long gid = (long long)blockIdx.x * blockDim.x + threadIdx.x;
  if (gid >= (long long)Bq * Eq * 4) return;
  int b   = (int)(gid / (Eq * 4));
  int rem = (int)(gid % (Eq * 4));
  int e = rem >> 2;
  int q = rem & 3;
  int row = ei[(size_t)b * 2 * Eq + e];
  int col = ei[(size_t)b * 2 * Eq + Eq + e];
  float w = ew[(size_t)b * Eq + e] * dinv[(size_t)b * Nq + row] * dinv[(size_t)b * Nq + col];
  const float* src = xw  + ((size_t)b * Nq + row) * HIDq + q * 16;
  float*       dst = agg + ((size_t)b * Nq + col) * HIDq + q * 16;
#pragma unroll
  for (int c = 0; c < 16; ++c) atomicAdd(&dst[c], src[c] * w);
}

// ==========================================================
// K10: h = relu(LN(agg + bn)) in place; bn = hn0[b, 4096+c]
// ==========================================================
__global__ void k_ln_relu(float* __restrict__ agg, const float* __restrict__ hn0,
                          const float* __restrict__ gam, const float* __restrict__ bet) {
  int node = blockIdx.x * blockDim.x + threadIdx.x;  // b*Nq + i
  if (node >= Bq * Nq) return;
  int b = node / Nq;
  const float* bn = hn0 + (size_t)b * H0q + HIDq * INq;
  float* row = agg + (size_t)node * HIDq;
  float s = 0.f, s2 = 0.f;
  for (int c = 0; c < HIDq; ++c) {
    float v = row[c] + bn[c];
    s += v; s2 += v * v;
  }
  float mu  = s * (1.f / HIDq);
  float var = s2 * (1.f / HIDq) - mu * mu;
  float inv = rsqrtf(var + 1e-5f);
  for (int c = 0; c < HIDq; ++c) {
    float v = (row[c] + bn[c] - mu) * inv * gam[c] + bet[c];
    row[c] = v > 0.f ? v : 0.f;
  }
}

// ==========================================================
// K14: layer-2 xw2[b,i,o] = dot(h1[b,i,:], Wn1[b,o,:])
// ==========================================================
__global__ void k_xw2(const float* __restrict__ h1, const float* __restrict__ hn1,
                      float* __restrict__ xw2) {
  int idx = blockIdx.x * blockDim.x + threadIdx.x;
  if (idx >= Bq * Nq * OUTq) return;
  int o = idx & 1;
  int i = (idx >> 1) & (Nq - 1);
  int b = idx >> 10;
  const float* hr = h1  + ((size_t)b * Nq + i) * HIDq;
  const float* wr = hn1 + (size_t)b * H1q + (size_t)o * HIDq;
  float acc = 0.f;
  for (int c = 0; c < HIDq; ++c) acc += hr[c] * wr[c];
  xw2[idx] = acc;
}

// ==========================================================
// K15: out init = xw2*dinv^2 + b'  (self loop + bias)
// ==========================================================
__global__ void k_out_init(const float* __restrict__ xw2, const float* __restrict__ dinv,
                           const float* __restrict__ hn1, float* __restrict__ out) {
  int idx = blockIdx.x * blockDim.x + threadIdx.x;
  if (idx >= Bq * Nq * OUTq) return;
  int o = idx & 1;
  int node = idx >> 1;               // b*Nq + i
  int b = node / Nq;
  float d = dinv[node];
  out[idx] = xw2[idx] * d * d + hn1[(size_t)b * H1q + OUTq * HIDq + o];
}

// ==========================================================
// K16: edge scatter (2 channels): 1 thread/edge
// ==========================================================
__global__ void k_edge_scatter2(const int* __restrict__ ei, const float* __restrict__ ew,
                                const float* __restrict__ dinv, const float* __restrict__ xw2,
                                float* __restrict__ out) {
  int idx = blockIdx.x * blockDim.x + threadIdx.x;
  if (idx >= Bq * Eq) return;
  int b = idx / Eq, e = idx % Eq;
  int row = ei[(size_t)b * 2 * Eq + e];
  int col = ei[(size_t)b * 2 * Eq + Eq + e];
  float w = ew[(size_t)b * Eq + e] * dinv[(size_t)b * Nq + row] * dinv[(size_t)b * Nq + col];
  atomicAdd(&out[((size_t)b * Nq + col) * OUTq + 0], xw2[((size_t)b * Nq + row) * OUTq + 0] * w);
  atomicAdd(&out[((size_t)b * Nq + col) * OUTq + 1], xw2[((size_t)b * Nq + row) * OUTq + 1] * w);
}

// ==========================================================
extern "C" void kernel_launch(void* const* d_in, const int* in_sizes, int n_in,
                              void* d_out, int out_size, void* d_ws, size_t ws_size,
                              hipStream_t stream) {
  const float* x    = (const float*)d_in[0];
  const int*   ei   = (const int*)  d_in[1];
  const float* ew   = (const float*)d_in[2];
  const float* p0   = (const float*)d_in[3];
  const float* p1   = (const float*)d_in[4];
  const float* W0   = (const float*)d_in[5];
  const float* b0   = (const float*)d_in[6];
  const float* W1   = (const float*)d_in[7];
  const float* b1   = (const float*)d_in[8];
  const float* gam  = (const float*)d_in[9];
  const float* bet  = (const float*)d_in[10];
  const float* Wih0 = (const float*)d_in[11];
  const float* Whh0 = (const float*)d_in[12];
  const float* bih0 = (const float*)d_in[13];
  const float* bhh0 = (const float*)d_in[14];
  const float* Wih1 = (const float*)d_in[15];
  const float* Whh1 = (const float*)d_in[16];
  const float* bih1 = (const float*)d_in[17];
  const float* bhh1 = (const float*)d_in[18];

  float* ws = (float*)d_ws;
  float*  hinit0 = ws + O_HINIT0;
  float*  hinit1 = ws + O_HINIT1;
  float*  gh0    = ws + O_GH0;
  float*  gh1    = ws + O_GH1;
  __bf16* Zbf    = (__bf16*)(ws + O_ZBF);
  float*  GI0    = ws + O_GI0;
  float*  hn0    = ws + O_HN0;
  __bf16* wbf    = (__bf16*)(ws + O_WBF);
  float*  dinv   = ws + O_DINV;
  float*  xw1    = ws + O_XW1;
  float*  agg1   = ws + O_AGG1;   // becomes h1 in place after LN+ReLU
  float*  GI1    = ws + O_GI1;
  float*  hn1    = ws + O_HN1;
  float*  xw2    = ws + O_XW2;
  float*  out    = (float*)d_out;

  // 1. static GRU hidden states
  k_build_hinit<<<(H0q + 255) / 256, 256, 0, stream>>>(W0, b0, W1, b1, hinit0, hinit1);
  // 2. scenario-invariant gh = Whh @ h_init + bhh (hoisted out of the batch)
  k_matvec_rows<<<(3 * H0q * 32 + 255) / 256, 256, 0, stream>>>(Whh0, hinit0, bhh0, gh0, 3 * H0q, H0q);
  k_matvec_rows<<<(3 * H1q * 32 + 255) / 256, 256, 0, stream>>>(Whh1, hinit1, bhh1, gh1, 3 * H1q, H1q);
  // 3. summarize on x -> Z (64 x 4096, bf16)
  k_summarize<<<Bq, Nq, 0, stream>>>(x, p0, Zbf);
  // 4. big WMMA GEMM: GI0 = Wih0 @ Z^T + bih0 (M=12480, K=4096, N=64; A read once)
  k_gemm_wmma<<<98, 256, 0, stream>>>(Wih0, Zbf, bih0, GI0, 3 * H0q);
  // 5. GRU0 elementwise -> hn0 (B x 4160) + bf16 weight copy
  k_gru_combine<<<(Bq * H0q + 255) / 256, 256, 0, stream>>>(GI0, gh0, hinit0, hn0, wbf, H0q, HIDq * INq);
  // 6. degree -> dinv (shared by both GCN layers)
  k_deg_init    <<<(Bq * Nq + 255) / 256, 256, 0, stream>>>(dinv);
  k_deg_edges   <<<(Bq * Eq + 255) / 256, 256, 0, stream>>>(ei, ew, dinv);
  k_deg_finalize<<<(Bq * Nq + 255) / 256, 256, 0, stream>>>(dinv);
  // 7. layer-1 xw = x @ Wn^T (batched WMMA, bf16 weights)
  k_xw1_wmma<<<dim3(Bq, 4), 256, 0, stream>>>(x, wbf, xw1);
  // 8-9. aggregate (self loop init + edge atomics)
  k_agg_init64    <<<(Bq * Nq * HIDq + 255) / 256, 256, 0, stream>>>(xw1, dinv, agg1);
  k_edge_scatter64<<<(int)(((long long)Bq * Eq * 4 + 255) / 256), 256, 0, stream>>>(ei, ew, dinv, xw1, agg1);
  // 10. +bias, LayerNorm, ReLU (in place -> h1)
  k_ln_relu<<<(Bq * Nq + 255) / 256, 256, 0, stream>>>(agg1, hn0, gam, bet);
  // 11. summarize on h1 -> Z (reuse)
  k_summarize<<<Bq, Nq, 0, stream>>>(agg1, p1, Zbf);
  // 12. WMMA GEMM: GI1 = Wih1 @ Z^T + bih1 (M=390 -> 25 m-tiles)
  k_gemm_wmma<<<4, 256, 0, stream>>>(Wih1, Zbf, bih1, GI1, 3 * H1q);
  // 13. GRU1 elementwise -> hn1 (B x 130), no bf16 copy
  k_gru_combine<<<(Bq * H1q + 255) / 256, 256, 0, stream>>>(GI1, gh1, hinit1, hn1, (__bf16*)nullptr, H1q, 0);
  // 14. layer-2 xw2 = h1 @ Wn1^T (N=2, plain VALU)
  k_xw2<<<(Bq * Nq * OUTq + 255) / 256, 256, 0, stream>>>(agg1, hn1, xw2);
  // 15-16. final aggregation directly into d_out
  k_out_init     <<<(Bq * Nq * OUTq + 255) / 256, 256, 0, stream>>>(xw2, dinv, hn1, out);
  k_edge_scatter2<<<(Bq * Eq + 255) / 256, 256, 0, stream>>>(ei, ew, dinv, xw2, out);
}
